// LorentzAttention_79714593013801
// MI455X (gfx1250) — matrix operand order repaired and checked by reading, not compiled, not run on hardware
//
#include <hip/hip_runtime.h>
#include <hip/hip_bf16.h>
#include <math.h>

// ---------------------------------------------------------------------------
// Types for CDNA5 WMMA (wave32, 16x16x32 f16 -> f32)
// ---------------------------------------------------------------------------
typedef __attribute__((ext_vector_type(16))) _Float16 v16h;
typedef __attribute__((ext_vector_type(8)))  float    v8f;
typedef __attribute__((ext_vector_type(4)))  unsigned int u32x4;
typedef __attribute__((ext_vector_type(8)))  int          i32x8;
typedef __attribute__((ext_vector_type(4)))  int          i32x4;

#define NN   512
#define DDIM 128
#define HH   4
#define DHH  32
#define PDD  64
#define BI   16
#define BJ   16
#define NJT  (NN / BJ)

// Enable the Tensor Data Mover path if the builtin exists (arity differs by
// toolchain: ROCm7.2/clang-22 = 5 args, upstream clang-23 = 6 args).
#ifndef ATHENA_TDM
#if defined(__has_builtin)
#if __has_builtin(__builtin_amdgcn_tensor_load_to_lds)
#define ATHENA_TDM 1
#endif
#endif
#endif
#ifndef ATHENA_TDM
#define ATHENA_TDM 0
#endif

__device__ inline v8f wmma16(v16h a, v16h b, v8f c) {
  return __builtin_amdgcn_wmma_f32_16x16x32_f16(false, a, false, b, (short)0, c,
                                                false, false);
}

struct F16x16 { uint4 lo; uint4 hi; };  // 32B, bit-castable to v16h

// A-matrix fragment (16x32 f16) from ROW-MAJOR storage.
// lane l holds row M=l&15; its 16 halves are K = khalf..khalf+15
// (khalf = (l>=16)?16:0) -> one aligned 32-byte run
// => 2 x b128 loads instead of 16 x u16.
__device__ inline v16h load_frag_a_rm(const _Float16* src, int ld, int row0,
                                      int k0, int lane) {
  const int m = lane & 15;
  const int khalf = (lane >> 4) << 4;
  const _Float16* r = src + (row0 + m) * ld + k0 + khalf;
  F16x16 t;
  t.lo = *(const uint4*)(r);
  t.hi = *(const uint4*)(r + 8);
  return __builtin_bit_cast(v16h, t);
}

// Fragment-linear storage: [slot][lane][16 halves] -- each lane's fragment is
// 32 contiguous, 32B-aligned bytes => 2 x b128 loads.
__device__ inline v16h load_frag_lin(const _Float16* base, int slot, int lane) {
  const _Float16* r = base + (slot * 32 + lane) * 16;
  F16x16 t;
  t.lo = *(const uint4*)(r);
  t.hi = *(const uint4*)(r + 8);
  return __builtin_bit_cast(v16h, t);
}

// C fragment (16x16 f32): VGPR r, lane l -> M = r + ((l>=16)?8:0), N = l&15
__device__ inline void store_frag_c(float* dst, int ld, int m0, int n0, v8f c,
                                    int lane) {
  const int n  = lane & 15;
  const int mr = (lane >> 4) << 3;
#pragma unroll
  for (int r = 0; r < 8; ++r) dst[(m0 + mr + r) * ld + n0 + n] = c[r];
}

// layer_norm(+bias) then SiLU over a 64-row
__device__ inline void ln_silu_row(const float* src, const float* bias,
                                   const float* g, const float* be,
                                   float* out64) {
  float x[64];
  float mu = 0.f;
#pragma unroll
  for (int c = 0; c < 64; ++c) {
    float v = bias ? (src[c] + bias[c]) : src[c];
    x[c] = v;
    mu += v;
  }
  mu *= 0.015625f;
  float va = 0.f;
#pragma unroll
  for (int c = 0; c < 64; ++c) {
    float d = x[c] - mu;
    va += d * d;
  }
  const float inv = rsqrtf(va * 0.015625f + 1e-5f);
#pragma unroll
  for (int c = 0; c < 64; ++c) {
    float y = (x[c] - mu) * inv * g[c] + be[c];
    out64[c] = y / (1.f + __expf(-y));
  }
}

__device__ inline float psi_f(float x) { return copysignf(log1pf(fabsf(x)), x); }

// ---------------------------------------------------------------------------
// Kernel A0: pack Wqkv (K=128 x N=384 f32) into f16 B-fragment-linear layout.
// slot = nt*4+kk ; lane n=l&15, khi=(l>=16)*16 ; element e -> K=kk*32+khi+e.
// ---------------------------------------------------------------------------
__global__ void k_pack_wqkv(const float* __restrict__ src,
                            _Float16* __restrict__ dst) {
  int s = blockIdx.x * blockDim.x + threadIdx.x;  // 49152 total
  if (s >= 128 * 384) return;
  const int e = s & 15, lane = (s >> 4) & 31, slot = s >> 9;
  const int kk = slot & 3, nt = slot >> 2;
  const int K = kk * 32 + (lane >> 4) * 16 + e;
  const int col = nt * 16 + (lane & 15);
  dst[s] = (_Float16)src[K * 384 + col];
}

// ---------------------------------------------------------------------------
// Kernel A1: layer_norm(feats, gamma, 0) -> f16 row-major
// ---------------------------------------------------------------------------
__global__ void k_ln_feats(const float* __restrict__ feats,
                           const float* __restrict__ gamma,
                           _Float16* __restrict__ xh) {
  const int n = blockIdx.x;
  const int d = threadIdx.x;  // 128
  __shared__ float red[128];
  float v = feats[n * DDIM + d];
  red[d] = v;
  __syncthreads();
  for (int s = 64; s > 0; s >>= 1) {
    if (d < s) red[d] += red[d + s];
    __syncthreads();
  }
  const float mean = red[0] * (1.f / 128.f);
  __syncthreads();
  float dv = v - mean;
  red[d] = dv * dv;
  __syncthreads();
  for (int s = 64; s > 0; s >>= 1) {
    if (d < s) red[d] += red[d + s];
    __syncthreads();
  }
  const float var = red[0] * (1.f / 128.f);
  xh[n * DDIM + d] = (_Float16)((v - mean) * rsqrtf(var + 1e-5f) * gamma[d]);
}

// ---------------------------------------------------------------------------
// Kernel A2: qkv = x @ Wqkv via WMMA (M=512, K=128, N=384)
// ---------------------------------------------------------------------------
__global__ __launch_bounds__(32) void k_qkv_gemm(const _Float16* __restrict__ xh,
                                                 const _Float16* __restrict__ wf,
                                                 float* __restrict__ qkvF) {
  const int lane = threadIdx.x & 31;
  const int mt = blockIdx.x, nt = blockIdx.y;
  v8f c = {};
#pragma unroll
  for (int kk = 0; kk < 4; ++kk) {
    v16h a = load_frag_a_rm(xh, DDIM, mt * 16, kk * 32, lane);
    v16h b = load_frag_lin(wf, nt * 4 + kk, lane);
    c = wmma16(a, b, c);
  }
  store_frag_c(qkvF, 3 * HH * DHH, mt * 16, nt * 16, c, lane);
}

// ---------------------------------------------------------------------------
// Kernel A3: split qkv, l2norm q/k; outputs:
//   qF: q as A-fragment-linear  [h][it][lane][16]  (f16)
//   kF: k as B-fragment-linear  [h][jt][lane][16]  (f16)
//   vF: v row-major             [h][n][d]          (f32)
// ---------------------------------------------------------------------------
__global__ void k_split_qkv(const float* __restrict__ qkvF,
                            _Float16* __restrict__ qF, _Float16* __restrict__ kF,
                            float* __restrict__ vF) {
  const int n = blockIdx.x;
  const int d = threadIdx.x;  // 128
  const int h = d >> 5, dd = d & 31;
  __shared__ float sq[128], sk[128], nrm[8];
  const float qv = qkvF[n * 384 + d];
  const float kv = qkvF[n * 384 + 128 + d];
  const float vv = qkvF[n * 384 + 256 + d];
  sq[d] = qv * qv;
  sk[d] = kv * kv;
  __syncthreads();
  if (d < 8) {
    const float* s = (d < 4) ? sq : sk;
    const int hh = d & 3;
    float acc = 0.f;
    for (int j = 0; j < 32; ++j) acc += s[hh * 32 + j];
    nrm[d] = fmaxf(sqrtf(acc), 1e-12f);
  }
  __syncthreads();
  {  // q -> A-fragment-linear
    const int it = n >> 4, m = n & 15;
    const int lanehi = (dd >> 4) & 1;
    const int e = dd & 15;
    const int lane = m + 16 * lanehi;
    qF[((h * 32 + it) * 32 + lane) * 16 + e] = (_Float16)(qv / nrm[h]);
  }
  {  // k -> B-fragment-linear
    const int jt = n >> 4, ncol = n & 15;
    const int lane = ncol + 16 * (dd >> 4);
    kF[((h * 32 + jt) * 32 + lane) * 16 + (dd & 15)] =
        (_Float16)(kv / nrm[4 + h]);
  }
  vF[h * NN * DHH + n * DHH + dd] = vv;
}

// ---------------------------------------------------------------------------
// Kernel B: fused pairwise MLP (WMMA) + qk (WMMA) + dual online softmax +
//           value/positional/coordinate accumulation + output projections.
// ---------------------------------------------------------------------------
// LDS layout (bytes)
#define OFF_TMPF   0        // 256*64 f32  = 65536 (also epilogue scratch)
#define OFF_HA     65536    // 256*64 f16  = 32768
#define OFF_W2     98304    // 64*64 f16 fragment-linear = 8192
#define OFF_W3     106496   // 64*64 f16 fragment-linear = 8192
#define OFF_S      114688   // 256*4 f32
#define OFF_P      118784   // 4*256 f32
#define OFF_CW     122880   // 256*4 f32
#define OFF_SG     126976   // 256*4 f32
#define OFF_RELN   131072   // 256*4 f32
#define OFF_RS     135168   // 64 f32
#define OFF_LS     135424   // 64 f32
#define OFF_RELOUT 135680   // 16*4*4 f32
#define OFF_PRM    136704   // 1120 f32 -> ends 141184
#define OFF_STAGE  141312   // TDM staging: 2 * 64*64 f32 = 32768
#define SMEM_BYTES 174080

__global__ __launch_bounds__(256) void k_lorentz_fused(
    const float* __restrict__ coors, const float* __restrict__ cmW1,
    const float* __restrict__ cmW2, const float* __restrict__ gW,
    const float* __restrict__ gb, const float* __restrict__ cscale,
    const float* __restrict__ comb, const float* __restrict__ dW0,
    const float* __restrict__ dW1, const float* __restrict__ dW2,
    const float* __restrict__ db0, const float* __restrict__ db1,
    const float* __restrict__ db2, const float* __restrict__ g0,
    const float* __restrict__ g1, const float* __restrict__ g2,
    const float* __restrict__ be0, const float* __restrict__ be1,
    const float* __restrict__ be2, const float* __restrict__ qkWp,
    const float* __restrict__ qkbp, const float* __restrict__ vhW,
    const float* __restrict__ vhB, const float* __restrict__ Wout,
    const float* __restrict__ bout, const _Float16* __restrict__ qF,
    const _Float16* __restrict__ kF, const float* __restrict__ vF,
    float* __restrict__ outNode, float* __restrict__ outCoors) {
  extern __shared__ char smem[];
  float*    tmpF = (float*)(smem + OFF_TMPF);
  _Float16* hA   = (_Float16*)(smem + OFF_HA);
  _Float16* W2f  = (_Float16*)(smem + OFF_W2);
  _Float16* W3f  = (_Float16*)(smem + OFF_W3);
  float* S    = (float*)(smem + OFF_S);
  float* P    = (float*)(smem + OFF_P);
  float* CW   = (float*)(smem + OFF_CW);
  float* SG   = (float*)(smem + OFF_SG);
  float* RELN = (float*)(smem + OFF_RELN);
  float* RS   = (float*)(smem + OFF_RS);
  float* LSH  = (float*)(smem + OFF_LS);
  float* RELO = (float*)(smem + OFF_RELOUT);
  float* prm  = (float*)(smem + OFF_PRM);
  float* pW1 = prm;          // 128
  float* pb1 = prm + 128;  float* pb2 = prm + 192; float* pb3 = prm + 256;
  float* pg1 = prm + 320;  float* pg2 = prm + 384; float* pg3 = prm + 448;
  float* pe1 = prm + 512;  float* pe2 = prm + 576; float* pe3 = prm + 640;
  float* pqkW = prm + 704;   // 256
  float* pqkb = prm + 960;   // 4
  float* pcm1 = prm + 964;   // 64
  float* pcm2 = prm + 1028;  // 64
  float* pgW  = prm + 1092;  // 16
  float* pgb  = prm + 1108;  // 4
  float* pcmb = prm + 1112;  // 4

  const int t = threadIdx.x;
  const int wave = t >> 5, lane = t & 31;
  const int i0 = blockIdx.x * BI;

  // ------- Tensor Data Mover: stage dpb_W[1]/dpb_W[2] (64x64 f32) in LDS ----
#if ATHENA_TDM
  if (t == 0) {
#pragma unroll
    for (int Lw = 0; Lw < 2; ++Lw) {
      const float* gsrc = Lw ? dW2 : dW1;
      const unsigned long long ga = (unsigned long long)(uintptr_t)gsrc;
      u32x4 d0;
      d0[0] = 1u;                                      // count=1
      d0[1] = (unsigned)(OFF_STAGE + Lw * 16384);      // lds_addr
      d0[2] = (unsigned)(ga & 0xffffffffu);            // global_addr lo
      d0[3] = (unsigned)((ga >> 32) & 0x01ffffffu) | (2u << 30);  // type=2
      i32x8 d1 = {};
      d1[0] = (int)(2u << 16);   // data_size = 4B
      d1[1] = (int)(64u << 16);  // tensor_dim0 = 64 (lo16)
      d1[2] = (int)(64u << 16);  // tensor_dim1 = 64 (lo16)
      d1[3] = (int)(64u << 16);  // tile_dim0 = 64
      d1[4] = 64;                // tile_dim1 = 64
      d1[5] = 64;                // tensor_dim0_stride = 64 (lo32)
      i32x4 z4 = {};
#if __clang_major__ >= 23
      i32x8 z8 = {};
      __builtin_amdgcn_tensor_load_to_lds(d0, d1, z4, z4, z8, 0);
#else
      __builtin_amdgcn_tensor_load_to_lds(d0, d1, z4, z4, 0);
#endif
    }
    __builtin_amdgcn_s_wait_tensorcnt((short)0);
  }
  __syncthreads();
  const float* w2src = (const float*)(smem + OFF_STAGE);
  const float* w3src = (const float*)(smem + OFF_STAGE + 16384);
#else
  const float* w2src = dW1;
  const float* w3src = dW2;
#endif

  // ------- pack MLP weights into B-fragment-linear f16 (slot = nt*2+kk) -----
  for (int s = t; s < 4096; s += 256) {
    const int e = s & 15, lp = (s >> 4) & 31, slot = s >> 9;
    const int kk = slot & 1, nt = slot >> 1;
    const int K = kk * 32 + (lp >> 4) * 16 + e;
    const int col = nt * 16 + (lp & 15);
    W2f[s] = (_Float16)w2src[K * 64 + col];
    W3f[s] = (_Float16)w3src[K * 64 + col];
  }
  // ------- small parameters into LDS -------
  if (t < 128) pW1[t] = dW0[t];
  if (t < 64) {
    pb1[t] = db0[t]; pb2[t] = db1[t]; pb3[t] = db2[t];
    pg1[t] = g0[t];  pg2[t] = g1[t];  pg3[t] = g2[t];
    pe1[t] = be0[t]; pe2[t] = be1[t]; pe3[t] = be2[t];
    pcm1[t] = cmW1[t];
    pcm2[t] = cmW2[t];
  }
  pqkW[t] = qkWp[t];  // 256 elements
  if (t < 16) pgW[t] = gW[t];
  if (t < 4) { pqkb[t] = qkbp[t]; pgb[t] = gb[t]; pcmb[t] = comb[t]; }

  // CDNA5 prefetch (global_prefetch_b8) of epilogue weights
  __builtin_prefetch(vhW + t * 32, 0, 0);
  __builtin_prefetch(Wout + t * 64, 0, 0);

  // q A-fragment (persistent): wave w<4 owns head h=w; one 32B load.
  v16h qfrag = {};
  if (wave < 4) qfrag = load_frag_lin(qF, wave * 32 + (i0 >> 4), lane);

  // persistent accumulators
  float m1 = -3.0e38f, l1 = 0.f;           // attention softmax (t<64)
  float m2 = -3.0e38f, l2 = 0.f;           // coordinate softmax (t<64)
  float aRel[4] = {0.f, 0.f, 0.f, 0.f};    // coord accum (t<64)
  float aOut[8];                           // attn @ v      (all threads)
  float aS[16];                            // attn @ hpos   (all threads)
#pragma unroll
  for (int z = 0; z < 8; ++z) aOut[z] = 0.f;
#pragma unroll
  for (int z = 0; z < 16; ++z) aS[z] = 0.f;

  const int u = t >> 2, hU = u >> 4, iU = u & 15;
  const int dbase = (t & 3) * 8, cbase = (t & 3) * 16;
  const int hS = t >> 4, iS = t & 15;
  const float csc = cscale[0];

  __syncthreads();

  for (int jt = 0; jt < NJT; ++jt) {
    const int j0 = jt * BJ;

    // ---- stage 1: pair features + layer1 MLP (thread-per-pair) ----
    {
      const int p = t, ii = p >> 4, jj = p & 15;
      const int gi = i0 + ii, gj = j0 + jj;
      const float a0 = coors[gi * 4 + 0], a1 = coors[gi * 4 + 1],
                  a2 = coors[gi * 4 + 2], a3 = coors[gi * 4 + 3];
      const float b0 = coors[gj * 4 + 0], b1 = coors[gj * 4 + 1],
                  b2 = coors[gj * 4 + 2], b3 = coors[gj * 4 + 3];
      const float r0 = a0 - b0, r1 = a1 - b1, r2 = a2 - b2, r3 = a3 - b3;
      const float nsq = r0 * r0 - r1 * r1 - r2 * r2 - r3 * r3;
      const float dsq = a0 * b0 - a1 * b1 - a2 * b2 - a3 * b3;
      const float f0 = -psi_f(nsq);
      const float f1 = psi_f(dsq);
      RELN[p * 4 + 0] = r0 * csc; RELN[p * 4 + 1] = r1 * csc;
      RELN[p * 4 + 2] = r2 * csc; RELN[p * 4 + 3] = r3 * csc;
      float t1[64];
#pragma unroll
      for (int c = 0; c < 64; ++c)
        t1[c] = f0 * pW1[c] + f1 * pW1[64 + c] + pb1[c];
      float o[64];
      ln_silu_row(t1, nullptr, pg1, pe1, o);
#pragma unroll
      for (int c = 0; c < 64; ++c) hA[p * 64 + c] = (_Float16)o[c];
    }
    __syncthreads();

    // ---- stage 2: qk WMMA (waves 0..3) + layer2 GEMM (all waves) ----
    if (wave < 4) {
      v16h kb = load_frag_lin(kF, wave * 32 + jt, lane);
      v8f cq = {};
      cq = wmma16(qfrag, kb, cq);
      const int n = lane & 15, mr = (lane >> 4) << 3;
#pragma unroll
      for (int r = 0; r < 8; ++r)
        S[((mr + r) * 16 + n) * 4 + wave] = cq[r] * 8.0f;  // SCALE
    }
#pragma unroll
    for (int mi = 0; mi < 2; ++mi) {
      const int mt = wave * 2 + mi;
#pragma unroll
      for (int nt = 0; nt < 4; ++nt) {
        v8f c = {};
        c = wmma16(load_frag_a_rm(hA, 64, mt * 16, 0, lane),
                   load_frag_lin(W2f, nt * 2 + 0, lane), c);
        c = wmma16(load_frag_a_rm(hA, 64, mt * 16, 32, lane),
                   load_frag_lin(W2f, nt * 2 + 1, lane), c);
        store_frag_c(tmpF, 64, mt * 16, nt * 16, c, lane);
      }
    }
    __syncthreads();

    // ---- stage 3: LN+SiLU layer2 -> hA ----
    {
      float o[64];
      ln_silu_row(tmpF + t * 64, pb2, pg2, pe2, o);
#pragma unroll
      for (int c = 0; c < 64; ++c) hA[t * 64 + c] = (_Float16)o[c];
    }
    __syncthreads();

    // ---- stage 4: layer3 GEMM ----
#pragma unroll
    for (int mi = 0; mi < 2; ++mi) {
      const int mt = wave * 2 + mi;
#pragma unroll
      for (int nt = 0; nt < 4; ++nt) {
        v8f c = {};
        c = wmma16(load_frag_a_rm(hA, 64, mt * 16, 0, lane),
                   load_frag_lin(W3f, nt * 2 + 0, lane), c);
        c = wmma16(load_frag_a_rm(hA, 64, mt * 16, 32, lane),
                   load_frag_lin(W3f, nt * 2 + 1, lane), c);
        store_frag_c(tmpF, 64, mt * 16, nt * 16, c, lane);
      }
    }
    __syncthreads();

    // ---- stage 5: final LN+SiLU -> hpos(f32), qk_pos, coord MLP ----
    {
      const int p = t;
      float o[64];
      ln_silu_row(tmpF + p * 64, pb3, pg3, pe3, o);
#pragma unroll
      for (int c = 0; c < 64; ++c) tmpF[p * 64 + c] = o[c];
      float qk4[4];
#pragma unroll
      for (int h = 0; h < 4; ++h) qk4[h] = pqkb[h];
#pragma unroll
      for (int c = 0; c < 64; ++c) {
        const float hv = o[c];
        qk4[0] += hv * pqkW[c * 4 + 0];
        qk4[1] += hv * pqkW[c * 4 + 1];
        qk4[2] += hv * pqkW[c * 4 + 2];
        qk4[3] += hv * pqkW[c * 4 + 3];
      }
      float w4[4];
#pragma unroll
      for (int h = 0; h < 4; ++h) {
        w4[h] = S[p * 4 + h] + qk4[h];
        S[p * 4 + h] = w4[h];
      }
      float cwv[4] = {0.f, 0.f, 0.f, 0.f};
#pragma unroll
      for (int ch = 0; ch < 16; ++ch) {
        float uu = w4[0] * pcm1[ch] + w4[1] * pcm1[16 + ch] +
                   w4[2] * pcm1[32 + ch] + w4[3] * pcm1[48 + ch];
        float ge = 0.5f * uu * (1.f + erff(uu * 0.70710678f));
        cwv[0] += ge * pcm2[ch * 4 + 0];
        cwv[1] += ge * pcm2[ch * 4 + 1];
        cwv[2] += ge * pcm2[ch * 4 + 2];
        cwv[3] += ge * pcm2[ch * 4 + 3];
      }
#pragma unroll
      for (int h = 0; h < 4; ++h) {
        CW[p * 4 + h] = cwv[h];
        float sgv = pgb[h] + w4[0] * pgW[h] + w4[1] * pgW[4 + h] +
                    w4[2] * pgW[8 + h] + w4[3] * pgW[12 + h];
        SG[p * 4 + h] = tanhf(sgv);
      }
    }
    __syncthreads();

    // ---- stage 6: dual online softmax (threads 0..63: one per (h,i)) ----
    if (t < 64) {
      float tmax = -3.0e38f;
#pragma unroll
      for (int jj = 0; jj < 16; ++jj)
        tmax = fmaxf(tmax, S[(iS * 16 + jj) * 4 + hS]);
      const float mn = fmaxf(m1, tmax);
      const float r = __expf(m1 - mn);
      float ps = 0.f;
#pragma unroll
      for (int jj = 0; jj < 16; ++jj) {
        float e = __expf(S[(iS * 16 + jj) * 4 + hS] - mn);
        P[hS * 256 + iS * 16 + jj] = e;
        ps += e;
      }
      l1 = l1 * r + ps;
      m1 = mn;
      RS[t] = r;
      float tmax2 = -3.0e38f;
#pragma unroll
      for (int jj = 0; jj < 16; ++jj)
        tmax2 = fmaxf(tmax2, CW[(iS * 16 + jj) * 4 + hS]);
      const float mn2 = fmaxf(m2, tmax2);
      const float r2 = __expf(m2 - mn2);
#pragma unroll
      for (int c = 0; c < 4; ++c) aRel[c] *= r2;
      float ps2 = 0.f;
#pragma unroll
      for (int jj = 0; jj < 16; ++jj) {
        const int p = iS * 16 + jj;
        float e2 = __expf(CW[p * 4 + hS] - mn2);
        ps2 += e2;
        float w = e2 * SG[p * 4 + hS];
#pragma unroll
        for (int c = 0; c < 4; ++c) aRel[c] += w * RELN[p * 4 + c];
      }
      l2 = l2 * r2 + ps2;
      m2 = mn2;
    }
    __syncthreads();

    // ---- stage 7: accumulate attn@v and attn@hpos (all 256 threads) ----
    {
      const float r = RS[u];
#pragma unroll
      for (int dd = 0; dd < 8; ++dd) aOut[dd] *= r;
#pragma unroll
      for (int cc = 0; cc < 16; ++cc) aS[cc] *= r;
      for (int jj = 0; jj < 16; ++jj) {
        const float pj = P[hU * 256 + iU * 16 + jj];
        const float* vrow = vF + hU * NN * DHH + (j0 + jj) * DHH + dbase;
#pragma unroll
        for (int dd = 0; dd < 8; ++dd) aOut[dd] += pj * vrow[dd];
        const float* hrow = tmpF + (iU * 16 + jj) * 64 + cbase;
#pragma unroll
        for (int cc = 0; cc < 16; ++cc) aS[cc] += pj * hrow[cc];
      }
    }
    __syncthreads();
  }

  // ======================= epilogue =======================
  if (t < 64) {
    LSH[t] = l1;
    const float inv2 = 1.f / l2;
#pragma unroll
    for (int c = 0; c < 4; ++c) RELO[(iS * 4 + c) * 4 + hS] = aRel[c] * inv2;
  }
  __syncthreads();
  {
    const float invl = 1.f / LSH[u];
#pragma unroll
    for (int cc = 0; cc < 16; ++cc)
      tmpF[u * 64 + cbase + cc] = aS[cc] * invl;             // sAcc [0..4095]
#pragma unroll
    for (int dd = 0; dd < 8; ++dd)
      tmpF[4096 + u * 32 + dbase + dd] = aOut[dd] * invl;    // oAttn
  }
  __syncthreads();
  // out[i][h*32+d] = oAttn + sAcc @ v_head_W + v_head_b
#pragma unroll
  for (int k = 0; k < 8; ++k) {
    const int z = t * 8 + k;
    const int i = z >> 7, col = z & 127;
    const int h = col >> 5, d = col & 31;
    const int u2 = h * 16 + i;
    float o = tmpF[4096 + u2 * 32 + d] + vhB[col];
    for (int c = 0; c < 64; ++c) o += tmpF[u2 * 64 + c] * vhW[c * 128 + col];
    tmpF[8192 + i * 128 + col] = o;
  }
  __syncthreads();
  // node_out = out @ W_out + b_out
#pragma unroll
  for (int k = 0; k < 8; ++k) {
    const int z = t * 8 + k;
    const int i = z >> 7, doo = z & 127;
    float acc = bout[doo];
    for (int c = 0; c < 128; ++c)
      acc += tmpF[8192 + i * 128 + c] * Wout[c * 128 + doo];
    outNode[(i0 + i) * DDIM + doo] = acc;
  }
  // coors_out = sum_h rel_out * combine
  if (t < 64) {
    const int i = t >> 2, c = t & 3;
    float v = 0.f;
#pragma unroll
    for (int h = 0; h < 4; ++h) v += RELO[(i * 4 + c) * 4 + h] * pcmb[h];
    outCoors[(i0 + i) * 4 + c] = v;
  }
}

// ---------------------------------------------------------------------------
// Host-side launch
// ---------------------------------------------------------------------------
extern "C" void kernel_launch(void* const* d_in, const int* in_sizes, int n_in,
                              void* d_out, int out_size, void* d_ws,
                              size_t ws_size, hipStream_t stream) {
  const float* feats  = (const float*)d_in[0];
  const float* coors  = (const float*)d_in[1];
  const float* gamma  = (const float*)d_in[2];
  const float* Wqkv   = (const float*)d_in[3];
  const float* W_out  = (const float*)d_in[4];
  const float* b_out  = (const float*)d_in[5];
  const float* cm_W1  = (const float*)d_in[6];
  const float* cm_W2  = (const float*)d_in[7];
  const float* gate_W = (const float*)d_in[8];
  const float* gate_b = (const float*)d_in[9];
  const float* cscale = (const float*)d_in[10];
  const float* comb   = (const float*)d_in[11];
  const float* dW0    = (const float*)d_in[12];
  const float* dW1    = (const float*)d_in[13];
  const float* dW2    = (const float*)d_in[14];
  const float* db0    = (const float*)d_in[15];
  const float* db1    = (const float*)d_in[16];
  const float* db2    = (const float*)d_in[17];
  const float* g0     = (const float*)d_in[18];
  const float* g1     = (const float*)d_in[19];
  const float* g2     = (const float*)d_in[20];
  const float* be0    = (const float*)d_in[21];
  const float* be1    = (const float*)d_in[22];
  const float* be2    = (const float*)d_in[23];
  const float* qkW    = (const float*)d_in[24];
  const float* qkb    = (const float*)d_in[25];
  const float* vhW    = (const float*)d_in[26];
  const float* vhB    = (const float*)d_in[27];

  char* ws = (char*)d_ws;
  _Float16* xh    = (_Float16*)(ws);                 // 512*128*2   = 131072
  _Float16* wqkvh = (_Float16*)(ws + 131072);        // 128*384*2   =  98304
  float*    qkvF  = (float*)(ws + 229376);           // 512*384*4   = 786432
  _Float16* qFr   = (_Float16*)(ws + 1015808);       // 4*512*32*2  = 131072
  _Float16* kFr   = (_Float16*)(ws + 1146880);       // 4*512*32*2  = 131072
  float*    vF    = (float*)(ws + 1277952);          // 4*512*32*4  = 262144

  float* outNode  = (float*)d_out;
  float* outCoors = outNode + NN * DDIM;

  k_pack_wqkv<<<192, 256, 0, stream>>>(Wqkv, wqkvh);
  k_ln_feats<<<NN, DDIM, 0, stream>>>(feats, gamma, xh);
  k_qkv_gemm<<<dim3(NN / 16, (3 * HH * DHH) / 16), 32, 0, stream>>>(xh, wqkvh,
                                                                    qkvF);
  k_split_qkv<<<NN, DDIM, 0, stream>>>(qkvF, qFr, kFr, vF);

  k_lorentz_fused<<<NN / BI, 256, SMEM_BYTES, stream>>>(
      coors, cm_W1, cm_W2, gate_W, gate_b, cscale, comb, dW0, dW1, dW2, db0,
      db1, db2, g0, g1, g2, be0, be1, be2, qkW, qkb, vhW, vhB, W_out, b_out,
      qFr, kFr, vF, outNode, outCoors);
}
